// OTTT_30940944400974
// MI455X (gfx1250) — matrix operand order repaired and checked by reading, not compiled, not run on hardware
//
#include <hip/hip_runtime.h>

// ---------- types for the WMMA builtin ----------
typedef __attribute__((ext_vector_type(16))) __bf16 v16bf;
typedef __attribute__((ext_vector_type(8)))  float  v8f;
typedef unsigned short u16;
typedef __attribute__((ext_vector_type(4)))  u16 v4u;
typedef __attribute__((ext_vector_type(8)))  u16 v8u;
typedef __attribute__((ext_vector_type(16))) u16 v16u;

#define LEAK   0.5f
#define NLEAK  0.5f
#define VTH    1.0f

// tiling
#define BM 128
#define BN 128
#define BK 32
#define LDA 40   // padded K-stride (halves) for A tile  (80B rows, 16B aligned)
#define LDB 40   // padded K-stride (halves) for B tile stored [N][K]

__device__ __forceinline__ u16 f2bf(float f) {
    // round-to-nearest-even f32 -> bf16
    unsigned int b = __float_as_uint(f);
    unsigned int r = b + 0x7FFFu + ((b >> 16) & 1u);
    return (u16)(r >> 16);
}

__device__ __forceinline__ v16bf load_frag_a(const u16* p) {
    // elements 0..7 = K {koff..koff+7}, elements 8..15 = K {koff+16..koff+23}
    v8u lo = *(const v8u*)(p);
    v8u hi = *(const v8u*)(p + 16);
    v16u f;
#pragma unroll
    for (int e = 0; e < 8; ++e) { f[e] = lo[e]; f[8 + e] = hi[e]; }
    return __builtin_bit_cast(v16bf, f);
}

__device__ __forceinline__ v16bf load_frag_b(const u16* p) {
    // elements 0..7 = K {koff..koff+7}, elements 8..15 = K {koff+16..koff+23}
    v8u lo = *(const v8u*)(p);
    v8u hi = *(const v8u*)(p + 16);
    v16u f;
#pragma unroll
    for (int e = 0; e < 8; ++e) { f[e] = lo[e]; f[8 + e] = hi[e]; }
    return __builtin_bit_cast(v16bf, f);
}

__global__ __launch_bounds__(256)
void ottt_gemm_lif(const float* __restrict__ x,
                   const float* __restrict__ u,
                   const float* __restrict__ W,
                   const float* __restrict__ bias,
                   float* __restrict__ u_next,
                   float* __restrict__ spikes,
                   int B, int Din, int Dout)
{
    __shared__ __align__(16) u16 sA[BM * LDA];  // x tile, [row][K] bf16
    __shared__ __align__(16) u16 sB[BN * LDB];  // W tile transposed, [N][K] bf16

    const int tid   = threadIdx.x;
    const int lane  = tid & 31;
    const int wave  = tid >> 5;
    const int waveM = (wave & 3) * 32;   // 4 waves along M
    const int waveN = (wave >> 2) * 64;  // 2 waves along N
    const int blockM = blockIdx.y * BM;
    const int blockN = blockIdx.x * BN;

    v8f acc[2][4];
    const v8f vzero = {0.f, 0.f, 0.f, 0.f, 0.f, 0.f, 0.f, 0.f};
#pragma unroll
    for (int i = 0; i < 2; ++i)
#pragma unroll
        for (int j = 0; j < 4; ++j) acc[i][j] = vzero;

    for (int k0 = 0; k0 < Din; k0 += BK) {
        // ---- stage A: 128x32 f32 -> bf16 LDS (4 float4 chunks / thread) ----
#pragma unroll
        for (int i = 0; i < 4; ++i) {
            int c   = tid + i * 256;          // 1024 float4 chunks
            int row = c >> 3;                 // 0..127
            int kc  = (c & 7) << 2;           // 0,4,..,28
            const float4 v = *(const float4*)(x + (size_t)(blockM + row) * Din + k0 + kc);
            v4u p; p[0] = f2bf(v.x); p[1] = f2bf(v.y); p[2] = f2bf(v.z); p[3] = f2bf(v.w);
            *(v4u*)(&sA[row * LDA + kc]) = p;
        }
        // ---- stage B: 32x128 f32 -> bf16 LDS transposed to [N][K] ----
#pragma unroll
        for (int i = 0; i < 4; ++i) {
            int c  = tid + i * 256;
            int kr = c >> 5;                  // 0..31
            int nc = (c & 31) << 2;           // 0,4,..,124
            const float4 v = *(const float4*)(W + (size_t)(k0 + kr) * Dout + blockN + nc);
            sB[(nc + 0) * LDB + kr] = f2bf(v.x);
            sB[(nc + 1) * LDB + kr] = f2bf(v.y);
            sB[(nc + 2) * LDB + kr] = f2bf(v.z);
            sB[(nc + 3) * LDB + kr] = f2bf(v.w);
        }
        __syncthreads();

        // ---- fragments per ISA 16-bit A(16x32)/B(32x16) layouts ----
        const int aoff = (lane >= 16) ? 8  : 0;   // halves
        const int boff = (lane >= 16) ? 8  : 0;   // halves
        v16bf afrag[2], bfrag[4];
#pragma unroll
        for (int tm = 0; tm < 2; ++tm)
            afrag[tm] = load_frag_a(&sA[(waveM + tm * 16 + (lane & 15)) * LDA + aoff]);
#pragma unroll
        for (int tn = 0; tn < 4; ++tn)
            bfrag[tn] = load_frag_b(&sB[(waveN + tn * 16 + (lane & 15)) * LDB + boff]);

#pragma unroll
        for (int tm = 0; tm < 2; ++tm)
#pragma unroll
            for (int tn = 0; tn < 4; ++tn)
                acc[tm][tn] = __builtin_amdgcn_wmma_f32_16x16x32_bf16(
                    /*neg_a=*/false, afrag[tm],
                    /*neg_b=*/false, bfrag[tn],
                    /*c_mod=*/(short)0, acc[tm][tn],
                    /*reuse_a=*/false, /*reuse_b=*/false);
        __syncthreads();
    }

    // ---- fused LIF epilogue from the C/D 16x16 f32 layout ----
    const int rsel = (lane >= 16) ? 8 : 0;
#pragma unroll
    for (int tm = 0; tm < 2; ++tm) {
#pragma unroll
        for (int tn = 0; tn < 4; ++tn) {
            const int col  = blockN + waveN + tn * 16 + (lane & 15);
            const float bc = bias[col];
            const int rb   = blockM + waveM + tm * 16 + rsel;
#pragma unroll
            for (int j = 0; j < 8; ++j) {
                const size_t idx = (size_t)(rb + j) * Dout + col;
                const float un = NLEAK * u[idx] + acc[tm][tn][j] + bc;
                const float s  = (un >= VTH) ? 1.0f : 0.0f;
                u_next[idx] = un - s * VTH;
                spikes[idx] = s;
            }
        }
    }
}

// a_hat_next = LEAK * a_hat + x   (pure bandwidth, float4 vectorized)
__global__ __launch_bounds__(256)
void ottt_trace(const float4* __restrict__ x,
                const float4* __restrict__ a_hat,
                float4* __restrict__ out, int n4)
{
    int i = blockIdx.x * blockDim.x + threadIdx.x;
    if (i < n4) {
        float4 xv = x[i], av = a_hat[i], r;
        r.x = fmaf(av.x, LEAK, xv.x);
        r.y = fmaf(av.y, LEAK, xv.y);
        r.z = fmaf(av.z, LEAK, xv.z);
        r.w = fmaf(av.w, LEAK, xv.w);
        out[i] = r;
    }
}

extern "C" void kernel_launch(void* const* d_in, const int* in_sizes, int n_in,
                              void* d_out, int out_size, void* d_ws, size_t ws_size,
                              hipStream_t stream)
{
    const float* x     = (const float*)d_in[0];
    const float* u     = (const float*)d_in[1];
    const float* a_hat = (const float*)d_in[2];
    const float* W     = (const float*)d_in[3];
    const float* b     = (const float*)d_in[4];

    const int Dout = in_sizes[4];
    const int Din  = in_sizes[3] / Dout;
    const int B    = in_sizes[0] / Din;

    float* u_next     = (float*)d_out;                         // [B, Dout]
    float* a_hat_next = u_next + (size_t)B * Dout;             // [B, Din]
    float* spikes     = a_hat_next + (size_t)B * Din;          // [B, Dout]

    dim3 grid(Dout / BN, B / BM);
    ottt_gemm_lif<<<grid, 256, 0, stream>>>(x, u, W, b, u_next, spikes, B, Din, Dout);

    const long n  = (long)B * Din;
    const int  n4 = (int)(n / 4);
    ottt_trace<<<(n4 + 255) / 256, 256, 0, stream>>>(
        (const float4*)x, (const float4*)a_hat, (float4*)a_hat_next, n4);
}
